// TeamShotGNN_41558103556528
// MI455X (gfx1250) — matrix operand order repaired and checked by reading, not compiled, not run on hardware
//
#include <hip/hip_runtime.h>
#include <math.h>

typedef _Float16 v16h __attribute__((ext_vector_type(16)));
typedef _Float16 v8h  __attribute__((ext_vector_type(8)));
typedef float    v8f  __attribute__((ext_vector_type(8)));

#define N_NODES 50000
#define N_EDGES 1600000
#define N_GRAPHS 64
#define HID 256
#define FIN 128

// ---------------- elementwise / prep kernels ----------------

__global__ void k_f32_to_f16(const float* __restrict__ in, _Float16* __restrict__ out, int n) {
    int tid = blockIdx.x * blockDim.x + threadIdx.x;
    if (tid < n) out[tid] = (_Float16)in[tid];
}

__global__ void k_deg(const long long* __restrict__ dst, float* __restrict__ deg, int e) {
    int tid = blockIdx.x * blockDim.x + threadIdx.x;
    if (tid < e) atomicAdd(&deg[(int)dst[tid]], 1.0f);
}

__global__ void k_dinv(float* __restrict__ deg, int n) {
    int tid = blockIdx.x * blockDim.x + threadIdx.x;
    if (tid < n) deg[tid] = rsqrtf(deg[tid] + 1.0f);
}

__global__ void k_edgeprep(const long long* __restrict__ ei, const float* __restrict__ dinv,
                           int* __restrict__ src32, int* __restrict__ dst32,
                           float* __restrict__ normv, int e) {
    int tid = blockIdx.x * blockDim.x + threadIdx.x;
    if (tid >= e) return;
    int s = (int)ei[tid];
    int d = (int)ei[(size_t)e + tid];
    src32[tid] = s;
    dst32[tid] = d;
    normv[tid] = dinv[s] * dinv[d];
}

// Repack row-major f32 weight [K,N] into WMMA B-fragment order (f16).
// Fragment (kt,nt16) holds 512 halves: half j of lane L is B[kt*32 + (j>>3)*16 + (L>>4)*8 + (j&7), nt16*16 + (L&15)].
__global__ void k_repack_w(const float* __restrict__ W, _Float16* __restrict__ Bf, int K, int N) {
    int tid = blockIdx.x * blockDim.x + threadIdx.x;
    if (tid >= K * N) return;
    int fragid = tid >> 9;
    int r      = tid & 511;
    int lane   = r >> 4;
    int j      = r & 15;
    int nfrag16 = N >> 4;
    int kt = fragid / nfrag16;
    int nt = fragid % nfrag16;
    int k  = (kt << 5) + ((j >> 3) << 4) + ((lane >> 4) << 3) + (j & 7);
    int c  = (nt << 4) + (lane & 15);
    Bf[tid] = (_Float16)W[(size_t)k * N + c];
}

// ---------------- WMMA GEMM: C[M,N] = A[M,K] (f16 row-major) x Bf (fragment-packed f16) ----------
// One wave computes a 16x128 tile (8 accumulators, A fragment reused 8x); 4 waves per block.
// Requires M % 16 == 0, N % 128 == 0, K % 32 == 0.
__global__ void k_wmma_gemm(const _Float16* __restrict__ A, const _Float16* __restrict__ Bf,
                            const float* __restrict__ bias, float* __restrict__ C,
                            _Float16* __restrict__ C16, int M, int N, int K, int relu) {
    int wid  = blockIdx.x * (blockDim.x >> 5) + (threadIdx.x >> 5);
    int lane = threadIdx.x & 31;
    int tilesN = N >> 7;
    int tiles  = (M >> 4) * tilesN;
    if (wid >= tiles) return;
    int mt = wid / tilesN;
    int nt = wid % tilesN;

    int arow = (mt << 4) + (lane & 15);
    int kb   = (lane >> 4) << 3;       // 0 or 8
    int nfrag16 = N >> 4;

    v8f acc[8];
    #pragma unroll
    for (int nn = 0; nn < 8; ++nn) acc[nn] = (v8f){0.f,0.f,0.f,0.f,0.f,0.f,0.f,0.f};

    const _Float16* arowp = A + (size_t)arow * K + kb;
    const size_t bkstep = ((size_t)nfrag16) << 9;           // halves per kt step of Bf
    const _Float16* bbase = Bf + (((size_t)(nt << 3)) << 9) + (lane << 4);

    for (int kt = 0; kt < (K >> 5); ++kt) {
        // A fragment: halves 0..7 = K[kb..kb+7], halves 8..15 = K[16+kb..16+kb+7]
        const _Float16* ap = arowp + (kt << 5);
        v8h alo = *(const v8h*)ap;
        v8h ahi = *(const v8h*)(ap + 16);
        v16h a;
        #pragma unroll
        for (int i = 0; i < 8; ++i) { a[i] = alo[i]; a[8 + i] = ahi[i]; }

        const _Float16* bp = bbase + (size_t)kt * bkstep;
        // prefetch next iteration's fragment streams (speculative; dropped if OOB)
        __builtin_prefetch(ap + 32, 0, 0);
        __builtin_prefetch(bp + bkstep, 0, 0);

        // batch all 8 B-fragment loads before the WMMA burst
        v16h b[8];
        #pragma unroll
        for (int nn = 0; nn < 8; ++nn) {
            v8h blo = *(const v8h*)(bp + ((size_t)nn << 9));
            v8h bhi = *(const v8h*)(bp + ((size_t)nn << 9) + 8);
            #pragma unroll
            for (int i = 0; i < 8; ++i) { b[nn][i] = blo[i]; b[nn][8 + i] = bhi[i]; }
        }
        #pragma unroll
        for (int nn = 0; nn < 8; ++nn) {
            acc[nn] = __builtin_amdgcn_wmma_f32_16x16x32_f16(
                /*neg_a=*/false, a, /*neg_b=*/false, b[nn],
                /*c_mod=*/(short)0, acc[nn], /*reuse_a=*/false, /*reuse_b=*/false);
        }
    }

    // C/D layout: VGPR r -> row = mt*16 + (lane>>4)*8 + r, col = base + (lane&15)
    int r0 = (mt << 4) + ((lane >> 4) << 3);
    int c0 = (nt << 7) + (lane & 15);
    #pragma unroll
    for (int nn = 0; nn < 8; ++nn) {
        int col = c0 + (nn << 4);
        float bv = bias ? bias[col] : 0.f;
        #pragma unroll
        for (int r = 0; r < 8; ++r) {
            float v = acc[nn][r] + bv;
            if (relu) v = fmaxf(v, 0.f);
            size_t idx = (size_t)(r0 + r) * N + col;
            C[idx] = v;
            if (C16) C16[idx] = (_Float16)v;
        }
    }
}

// ---------------- edge message: agg[dst] += h[src] * norm ----------------
// thread = (edge, float4 chunk); chunk fastest -> coalesced gather of one h-row per 64 threads.
__global__ void k_edge_msg(const int* __restrict__ src, const int* __restrict__ dst,
                           const float* __restrict__ normv, const float* __restrict__ h,
                           float* __restrict__ agg, long long total) {
    long long tid = (long long)blockIdx.x * blockDim.x + threadIdx.x;
    if (tid >= total) return;
    int c = (int)(tid & 63);         // HID/4 = 64 chunks
    int e = (int)(tid >> 6);
    int s = src[e];
    int d = dst[e];
    float n = normv[e];
    float4 v = ((const float4*)(h + (size_t)s * HID))[c];
    float* ap = agg + (size_t)d * HID + (c << 2);
    atomicAdd(ap + 0, v.x * n);
    atomicAdd(ap + 1, v.y * n);
    atomicAdd(ap + 2, v.z * n);
    atomicAdd(ap + 3, v.w * n);
}

// act = relu(agg + h*dinv^2 + b); written in-place into agg (f32) and to act16 (f16)
__global__ void k_node(float* __restrict__ agg, const float* __restrict__ hw,
                       const float* __restrict__ dinv, const float* __restrict__ bias,
                       _Float16* __restrict__ act16, int n) {
    int tid = blockIdx.x * blockDim.x + threadIdx.x;
    if (tid >= n) return;
    int row = tid >> 8;
    int col = tid & 255;
    float d = dinv[row];
    float v = agg[tid] + hw[tid] * d * d + bias[col];
    v = fmaxf(v, 0.f);
    agg[tid] = v;
    act16[tid] = (_Float16)v;
}

// ---------------- pooling ----------------

__global__ void k_count(const long long* __restrict__ batch, float* __restrict__ gcnt, int n) {
    int tid = blockIdx.x * blockDim.x + threadIdx.x;
    if (tid < n) atomicAdd(&gcnt[(int)batch[tid]], 1.0f);
}

__global__ void k_pool(const long long* __restrict__ batch, const float* __restrict__ act,
                       float* __restrict__ gbuf, int n) {
    int tid = blockIdx.x * blockDim.x + threadIdx.x;
    if (tid >= n) return;
    int node = tid >> 8;
    int f    = tid & 255;
    int g    = (int)batch[node];
    float v  = act[tid];
    atomicAdd(&gbuf[(size_t)g * 512 + f], v);
    // post-ReLU values are >= 0, buffer init 0 -> int-bit max == float max
    atomicMax((int*)&gbuf[(size_t)g * 512 + 256 + f], __float_as_int(v));
}

__global__ void k_poolfin(float* __restrict__ gbuf, const float* __restrict__ gcnt,
                          _Float16* __restrict__ g16, int n) {
    int tid = blockIdx.x * blockDim.x + threadIdx.x;
    if (tid >= n) return;
    int g = tid >> 9;
    int f = tid & 511;
    float v = gbuf[tid];
    if (f < 256) { v = v / fmaxf(gcnt[g], 1.0f); gbuf[tid] = v; }
    g16[tid] = (_Float16)v;
}

__global__ void k_head_out(const float* __restrict__ h2, const float* __restrict__ W6,
                           const float* __restrict__ b6, float* __restrict__ out) {
    int g = threadIdx.x;
    if (g >= N_GRAPHS) return;
    float acc = b6[0];
    for (int j = 0; j < 128; ++j) acc += h2[(size_t)g * 128 + j] * W6[j];
    out[g] = 1.0f / (1.0f + expf(-acc));
}

// ---------------- host launcher ----------------

extern "C" void kernel_launch(void* const* d_in, const int* in_sizes, int n_in,
                              void* d_out, int out_size, void* d_ws, size_t ws_size,
                              hipStream_t stream) {
    const float*     x     = (const float*)d_in[0];
    const long long* ei    = (const long long*)d_in[1];
    const long long* batch = (const long long*)d_in[2];
    const float* W1 = (const float*)d_in[3];  const float* b1 = (const float*)d_in[4];
    const float* W2 = (const float*)d_in[5];  const float* b2 = (const float*)d_in[6];
    const float* W3 = (const float*)d_in[7];  const float* b3 = (const float*)d_in[8];
    const float* W4 = (const float*)d_in[9];  const float* b4 = (const float*)d_in[10];
    const float* W5 = (const float*)d_in[11]; const float* b5 = (const float*)d_in[12];
    const float* W6 = (const float*)d_in[13]; const float* b6 = (const float*)d_in[14];
    float* out = (float*)d_out;

    // workspace carve-out (256B aligned slices)
    char* p = (char*)d_ws;
    auto carve = [&p](size_t bytes) -> char* {
        char* r = p;
        p += (bytes + 255) & ~(size_t)255;
        return r;
    };
    int*       src32 = (int*)carve((size_t)N_EDGES * 4);
    int*       dst32 = (int*)carve((size_t)N_EDGES * 4);
    float*     normv = (float*)carve((size_t)N_EDGES * 4);
    float*     dinv  = (float*)carve((size_t)N_NODES * 4);
    _Float16*  in16  = (_Float16*)carve((size_t)N_NODES * HID * 2);
    float*     hw    = (float*)carve((size_t)N_NODES * HID * 4);
    float*     agg   = (float*)carve((size_t)N_NODES * HID * 4);
    _Float16*  w1f   = (_Float16*)carve((size_t)FIN * HID * 2);
    _Float16*  w2f   = (_Float16*)carve((size_t)HID * HID * 2);
    _Float16*  w3f   = (_Float16*)carve((size_t)HID * HID * 2);
    _Float16*  w4f   = (_Float16*)carve((size_t)512 * HID * 2);
    _Float16*  w5f   = (_Float16*)carve((size_t)HID * 128 * 2);
    float*     gbuf  = (float*)carve((size_t)N_GRAPHS * 512 * 4);
    float*     gcnt  = (float*)carve((size_t)N_GRAPHS * 4);
    _Float16*  g16   = (_Float16*)carve((size_t)N_GRAPHS * 512 * 2);
    float*     h1g   = (float*)carve((size_t)N_GRAPHS * HID * 4);
    _Float16*  h1g16 = (_Float16*)carve((size_t)N_GRAPHS * HID * 2);
    float*     h2g   = (float*)carve((size_t)N_GRAPHS * 128 * 4);

    const int T = 256;

    // degree -> dinv, edge prep
    hipMemsetAsync(dinv, 0, (size_t)N_NODES * 4, stream);
    k_deg<<<(N_EDGES + T - 1) / T, T, 0, stream>>>(ei + N_EDGES, dinv, N_EDGES);
    k_dinv<<<(N_NODES + T - 1) / T, T, 0, stream>>>(dinv, N_NODES);
    k_edgeprep<<<(N_EDGES + T - 1) / T, T, 0, stream>>>(ei, dinv, src32, dst32, normv, N_EDGES);

    // weight repacks + x -> f16
    k_repack_w<<<(FIN * HID + T - 1) / T, T, 0, stream>>>(W1, w1f, FIN, HID);
    k_repack_w<<<(HID * HID + T - 1) / T, T, 0, stream>>>(W2, w2f, HID, HID);
    k_repack_w<<<(HID * HID + T - 1) / T, T, 0, stream>>>(W3, w3f, HID, HID);
    k_repack_w<<<(512 * HID + T - 1) / T, T, 0, stream>>>(W4, w4f, 512, HID);
    k_repack_w<<<(HID * 128 + T - 1) / T, T, 0, stream>>>(W5, w5f, HID, 128);
    k_f32_to_f16<<<((N_NODES * FIN) + T - 1) / T, T, 0, stream>>>(x, in16, N_NODES * FIN);

    // 3 GCN layers
    const float* biases[3] = {b1, b2, b3};
    const _Float16* wfs[3] = {w1f, w2f, w3f};
    const int Ks[3] = {FIN, HID, HID};
    const int tilesConv = (N_NODES / 16) * (HID / 128);      // 6250 waves
    const long long edgeTot = (long long)N_EDGES * (HID / 4);
    for (int L = 0; L < 3; ++L) {
        k_wmma_gemm<<<(tilesConv + 3) / 4, 128, 0, stream>>>(
            in16, wfs[L], nullptr, hw, nullptr, N_NODES, HID, Ks[L], 0);
        hipMemsetAsync(agg, 0, (size_t)N_NODES * HID * 4, stream);
        k_edge_msg<<<(unsigned)((edgeTot + T - 1) / T), T, 0, stream>>>(
            src32, dst32, normv, hw, agg, edgeTot);
        k_node<<<((N_NODES * HID) + T - 1) / T, T, 0, stream>>>(
            agg, hw, dinv, biases[L], in16, N_NODES * HID);
    }

    // pooling (agg holds final activations in f32)
    hipMemsetAsync(gbuf, 0, (size_t)N_GRAPHS * 512 * 4, stream);
    hipMemsetAsync(gcnt, 0, (size_t)N_GRAPHS * 4, stream);
    k_count<<<(N_NODES + T - 1) / T, T, 0, stream>>>(batch, gcnt, N_NODES);
    k_pool<<<((N_NODES * HID) + T - 1) / T, T, 0, stream>>>(batch, agg, gbuf, N_NODES * HID);
    k_poolfin<<<((N_GRAPHS * 512) + T - 1) / T, T, 0, stream>>>(gbuf, gcnt, g16, N_GRAPHS * 512);

    // MLP head: [64,512]@[512,256] relu -> [64,256]@[256,128] relu -> [64,128]@[128,1] sigmoid
    {
        int tiles1 = (N_GRAPHS / 16) * (HID / 128);  // 8 waves
        k_wmma_gemm<<<(tiles1 + 3) / 4, 128, 0, stream>>>(
            g16, w4f, b4, h1g, h1g16, N_GRAPHS, HID, 512, 1);
        int tiles2 = (N_GRAPHS / 16) * (128 / 128);  // 4 waves
        k_wmma_gemm<<<(tiles2 + 3) / 4, 128, 0, stream>>>(
            h1g16, w5f, b5, h2g, nullptr, N_GRAPHS, 128, HID, 1);
        k_head_out<<<1, 64, 0, stream>>>(h2g, W6, b6, out);
    }
}
